// Rvt_43336220017025
// MI455X (gfx1250) — matrix operand order repaired and checked by reading, not compiled, not run on hardware
//
#include <hip/hip_runtime.h>
#include <hip/hip_bf16.h>
#include <stdint.h>

// ---------- CDNA5 WMMA plumbing (wave32, 16x16x32 bf16, f32 accum) ----------
typedef __attribute__((ext_vector_type(16))) __bf16          v16bf;
typedef __attribute__((ext_vector_type(16))) unsigned short  v16us;
typedef __attribute__((ext_vector_type(8)))  float           v8f;
typedef __attribute__((ext_vector_type(4)))  unsigned int    u32x4;
typedef __attribute__((ext_vector_type(2)))  unsigned int    u32x2;
typedef __attribute__((ext_vector_type(4)))  float           f32x4;
typedef __attribute__((ext_vector_type(8)))  int             i32x8;
typedef __attribute__((ext_vector_type(4)))  int             i32x4;

#define LDST 40   // LDS row stride in ushorts (80B = 64B data + 16B pad)

#if defined(__has_builtin)
#if __has_builtin(__builtin_amdgcn_tensor_load_to_lds) && \
    __has_builtin(__builtin_amdgcn_s_wait_tensorcnt)
#define HAVE_TDM 1
#endif
#endif
#ifndef HAVE_TDM
#define HAVE_TDM 0
#endif

__device__ __forceinline__ unsigned short f2bf(float f) {
  unsigned int u = __float_as_uint(f);
  unsigned int r = u + 0x7FFFu + ((u >> 16) & 1u);   // round-to-nearest-even
  return (unsigned short)(r >> 16);
}
__device__ __forceinline__ unsigned int pack2bf(float a, float b) {
  return (unsigned int)f2bf(a) | ((unsigned int)f2bf(b) << 16);
}

#if HAVE_TDM
// TDM: DMA a 2D tile of 16-bit elements (tile0 elems/row -> 64B rows, tile1 rows,
// row stride stride0 elems) from global into LDS at lds_off, inserting 16B of
// LDS padding after each 64B row => 80B LDS pitch == LDST ushorts.
// D# layout per CDNA5 ISA ch.8 (group0 128b, group1 256b; groups 2/3/4 unused: 2D).
// This toolchain exposes the 6-arg builtin (g0, g1, g2, g3, g4, cpol).
__device__ __forceinline__ void tdm_load_2d(unsigned lds_off, const void* gptr,
                                            unsigned td0, unsigned td1,
                                            unsigned tile0, unsigned tile1,
                                            unsigned stride0) {
  unsigned long long ga = (unsigned long long)(uintptr_t)gptr;
  u32x4 g0;
  g0[0] = 1u;                                                  // count=1 (valid)
  g0[1] = lds_off;                                             // lds_addr (bytes)
  g0[2] = (unsigned)ga;                                        // global_addr lo
  g0[3] = (unsigned)((ga >> 32) & 0x01FFFFFFu) | (2u << 30);   // addr hi | type=2
  i32x8 g1;
  g1[0] = (int)((1u << 16) |        // data_size = 2 bytes
                (1u << 20) |        // pad_enable
                (3u << 22) |        // pad_interval: pad after 64B
                (3u << 25));        // pad_amount: 4 DWORDs (16B)
  g1[1] = (int)((td0 & 0xFFFFu) << 16);                              // tensor_dim0 lo
  g1[2] = (int)(((td0 >> 16) & 0xFFFFu) | ((td1 & 0xFFFFu) << 16));  // d0 hi | d1 lo
  g1[3] = (int)(((td1 >> 16) & 0xFFFFu) | ((tile0 & 0xFFFFu) << 16));// d1 hi | tile0
  g1[4] = (int)(tile1 & 0xFFFFu);                                    // tile1 | tile2=0
  g1[5] = (int)stride0;                                              // dim0 stride lo
  g1[6] = 0;
  g1[7] = 0;
  i32x4 gz = {0, 0, 0, 0};
  i32x8 gz8 = {0, 0, 0, 0, 0, 0, 0, 0};
  __builtin_amdgcn_tensor_load_to_lds(g0, g1, gz, gz, gz8, 0);
}
#endif

// A fragment (16x32 MxK bf16): lane -> M=row; per-lane K run = [16h..16h+15].
// LDS tile stored [M][K].
__device__ __forceinline__ v16us ld_frag_a(const unsigned short* base, int row, int lane) {
  const int hlf = (lane >> 4) & 1;
  const unsigned short* p = base + row * LDST;
  union { u32x4 q[2]; v16us v; } u;
  u.q[0] = *(const u32x4*)(p + 16 * hlf);
  u.q[1] = *(const u32x4*)(p + 16 * hlf + 8);
  return u.v;
}
// B fragment (32x16 KxN bf16): lane -> N=col; per-lane K run = [16h..16h+15].
// LDS tile stored transposed [N][K].
__device__ __forceinline__ v16us ld_frag_b(const unsigned short* base, int col0, int lane) {
  const int hlf = (lane >> 4) & 1;
  const unsigned short* p = base + (col0 + (lane & 15)) * LDST + 16 * hlf;
  union { u32x4 q[2]; v16us v; } u;
  u.q[0] = *(const u32x4*)(p);
  u.q[1] = *(const u32x4*)(p + 8);
  return u.v;
}
__device__ __forceinline__ v8f wmma_bf16(v16us a, v16us b, v8f c) {
  union { v16us u; v16bf h; } ua, ub;
  ua.u = a; ub.u = b;
  return __builtin_amdgcn_wmma_f32_16x16x32_bf16(false, ua.h, false, ub.h,
                                                 (short)0, c, false, false);
}

// ---------------- problem constants ----------------
// N=4, I=128, H=W=48, R=A=64, HEADS=4, HD=16
// P = 9216, HW = 2304, K(3x3) = 1152, D = 46*46 = 2116
// Layouts (position-major so K-dim is contiguous):
//   xh  [n*2304+pos][128] bf16     QKV [p][192] bf16 (q:0-63 k:64-127 v:128-191)
//   Gx  [p][256] f32 (i,f,g,o)     Abuf[p][64]  bf16

// ---------------- kernel 0: weight prep ----------------
__global__ void prep_kernel(
    const float* W_x, const float* W_q, const float* W_k, const float* W_v,
    const float* Wi_x, const float* Wf_x, const float* Wg_x, const float* Wo_x,
    const float* Wi_a, const float* Wf_a, const float* Wg_a, const float* Wo_a,
    const float* b_v, const float* b_i, const float* b_f, const float* b_g, const float* b_o,
    unsigned short* Wall, unsigned short* Wx, unsigned short* Wa, float* bias_all)
{
  const int T1 = 448 * 1152, T2 = 64 * 128, T3 = 256 * 64, T4 = 448;
  const float* xsrc[7] = {W_q, W_k, W_v, Wi_x, Wf_x, Wg_x, Wo_x};
  const float* asrc[4] = {Wi_a, Wf_a, Wg_a, Wo_a};
  const float* bsrc[4] = {b_i, b_f, b_g, b_o};
  const int total = T1 + T2 + T3 + T4;
  for (int idx = blockIdx.x * blockDim.x + threadIdx.x; idx < total;
       idx += gridDim.x * blockDim.x) {
    if (idx < T1) {
      int m = idx / 1152, k = idx - m * 1152;
      int tap = k >> 7, ci = k & 127;
      int blk = m >> 6, oc = m & 63;
      Wall[m * 1152 + k] = f2bf(xsrc[blk][(oc * 128 + ci) * 9 + tap]);   // k=(tap,ci)
    } else if (idx < T1 + T2) {
      int i = idx - T1;
      Wx[i] = f2bf(W_x[i]);
    } else if (idx < T1 + T2 + T3) {
      int i = idx - T1 - T2;
      int r = i >> 6, ac = i & 63;
      Wa[i] = f2bf(asrc[r >> 6][(r & 63) * 64 + ac]);
    } else {
      int m = idx - T1 - T2 - T3;
      float bv = 0.f;
      if (m >= 192)       bv = bsrc[(m - 192) >> 6][m & 63];
      else if (m >= 128)  bv = b_v[m - 128];
      bias_all[m] = bv;
    }
  }
}

// ---------------- kernel 1: proj_x 1x1 GEMM (64x128x9216) + h concat ----------------
__global__ void __launch_bounds__(256) proj_concat_kernel(
    const float* x_in, const float* h, const unsigned short* Wx, const float* b_x,
    unsigned short* xh)
{
  __shared__ __align__(16) unsigned short sa[64 * LDST];
  __shared__ __align__(16) unsigned short sb[128 * LDST];
  const int tid = threadIdx.x, lane = tid & 31, wave = tid >> 5;
  const int p0 = blockIdx.x * 128;         // 2304 % 128 == 0 -> one image per tile
  const int n = p0 / 2304;
  const int posb = p0 - n * 2304;
  const int mi = wave >> 1, pi0 = (wave & 1) * 4;
  const int hlf = (lane >> 4) & 1;

  v8f z8 = {0,0,0,0,0,0,0,0};
  v8f acc[4] = {z8, z8, z8, z8};

  for (int kc = 0; kc < 4; ++kc) {                       // K = 128
#if HAVE_TDM
    if (wave == 0)                                       // async weight tile DMA
      tdm_load_2d((unsigned)(uintptr_t)sa, (const void*)&Wx[kc * 32],
                  128, 64, 32, 64, 128);
#else
    { int m = tid >> 2, kg = tid & 3;
      *(u32x4*)&sa[m * LDST + kg * 8] =
          *(const u32x4*)&Wx[m * 128 + kc * 32 + kg * 8]; }
#endif
    for (int e = tid; e < 512; e += 256) {               // x_in: batched strided gather
      int col = e >> 2, kg = e & 3;
      float t[8];
#pragma unroll
      for (int u = 0; u < 8; ++u)
        t[u] = x_in[(n * 128 + kc * 32 + kg * 8 + u) * 2304 + posb + col];
      u32x4 v;
#pragma unroll
      for (int u = 0; u < 4; ++u) v[u] = pack2bf(t[2*u], t[2*u+1]);
      *(u32x4*)&sb[col * LDST + kg * 8] = v;
    }
#if HAVE_TDM
    if (wave == 0) __builtin_amdgcn_s_wait_tensorcnt(0);
#endif
    __syncthreads();
    v16us af = ld_frag_a(sa, mi * 16 + (lane & 15), lane);
#pragma unroll
    for (int pi = 0; pi < 4; ++pi) {
      v16us bf = ld_frag_b(sb, (pi0 + pi) * 16, lane);
      acc[pi] = wmma_bf16(af, bf, acc[pi]);
    }
    __syncthreads();
  }
  const int m8 = mi * 16 + hlf * 8;
#pragma unroll
  for (int pi = 0; pi < 4; ++pi) {                       // packed b128 epilogue
    int col = (pi0 + pi) * 16 + (lane & 15);
    u32x4 v;
#pragma unroll
    for (int u = 0; u < 4; ++u)
      v[u] = pack2bf(acc[pi][2*u] + b_x[m8 + 2*u], acc[pi][2*u+1] + b_x[m8 + 2*u+1]);
    *(u32x4*)&xh[(n * 2304 + posb + col) * 128 + m8] = v;
  }
  for (int e = tid; e < 1024; e += 256) {                // h -> channels 64..127
    int pos = e >> 3, cg = e & 7;
    float t[8];
#pragma unroll
    for (int u = 0; u < 8; ++u)
      t[u] = h[(n * 64 + cg * 8 + u) * 2304 + posb + pos];
    u32x4 v;
#pragma unroll
    for (int u = 0; u < 4; ++u) v[u] = pack2bf(t[2*u], t[2*u+1]);
    *(u32x4*)&xh[(n * 2304 + posb + pos) * 128 + 64 + cg * 8] = v;
  }
}

// ---------------- kernel 2: fused 3x3 conv, implicit GEMM 448x1152x9216 ----------------
__global__ void __launch_bounds__(256) conv448_kernel(
    const unsigned short* xh, const unsigned short* Wall, const float* bias_all,
    unsigned short* QKV, float* Gx)
{
  __shared__ __align__(16) unsigned short sa[64 * LDST];
  __shared__ __align__(16) unsigned short sb[128 * LDST];
  const int tid = threadIdx.x, lane = tid & 31, wave = tid >> 5;
  const int p0 = blockIdx.x * 128;
  const int m0 = blockIdx.y * 64;
  const int n = p0 / 2304;
  const int posb = p0 - n * 2304;
  const int mi = wave >> 1, pi0 = (wave & 1) * 4;
  const int hlf = (lane >> 4) & 1;
  const int kg = tid & 3;
  // hoisted im2col coordinates: this thread fills cols colA and colA+64
  const int colA = tid >> 2;
  const int yA = (posb + colA) / 48, xA = (posb + colA) - yA * 48;
  const int yB = (posb + colA + 64) / 48, xB = (posb + colA + 64) - yB * 48;
  const u32x4 zv4 = {0, 0, 0, 0};

  v8f z8 = {0,0,0,0,0,0,0,0};
  v8f acc[4] = {z8, z8, z8, z8};

  for (int kc = 0; kc < 36; ++kc) {                      // 9 taps x 4 chunks of 32
    int tap = kc >> 2, ci0 = (kc & 3) * 32;
    int dy = tap / 3 - 1, dx = tap % 3 - 1;
#if HAVE_TDM
    if (wave == 0)                                       // async weight tile DMA
      tdm_load_2d((unsigned)(uintptr_t)sa,
                  (const void*)&Wall[(size_t)m0 * 1152 + kc * 32],
                  1152, 448, 32, 64, 1152);
#else
    { int m = tid >> 2;
      *(u32x4*)&sa[m * LDST + kg * 8] =
          *(const u32x4*)&Wall[(m0 + m) * 1152 + kc * 32 + kg * 8]; }
#endif
    {                                                    // im2col: b128 per thread x2
      int yy = yA + dy, xx = xA + dx;
      u32x4 v = zv4;
      if ((unsigned)yy < 48u && (unsigned)xx < 48u)
        v = *(const u32x4*)&xh[(n * 2304 + yy * 48 + xx) * 128 + ci0 + kg * 8];
      *(u32x4*)&sb[colA * LDST + kg * 8] = v;
      yy = yB + dy; xx = xB + dx;
      u32x4 w = zv4;
      if ((unsigned)yy < 48u && (unsigned)xx < 48u)
        w = *(const u32x4*)&xh[(n * 2304 + yy * 48 + xx) * 128 + ci0 + kg * 8];
      *(u32x4*)&sb[(colA + 64) * LDST + kg * 8] = w;
    }
    if (kc + 1 < 36)                                     // -> global_prefetch_b8
      __builtin_prefetch(&Wall[(m0 + (tid >> 2)) * 1152 + (kc + 1) * 32], 0, 0);
#if HAVE_TDM
    if (wave == 0) __builtin_amdgcn_s_wait_tensorcnt(0);
#endif
    __syncthreads();
    v16us af = ld_frag_a(sa, mi * 16 + (lane & 15), lane);
#pragma unroll
    for (int pi = 0; pi < 4; ++pi) {
      v16us bf = ld_frag_b(sb, (pi0 + pi) * 16, lane);
      acc[pi] = wmma_bf16(af, bf, acc[pi]);
    }
    __syncthreads();
  }
  const int m8 = m0 + mi * 16 + hlf * 8;
  if (m0 < 192) {                                        // q/k/v -> bf16 [p][192]
#pragma unroll
    for (int pi = 0; pi < 4; ++pi) {
      int p = p0 + (pi0 + pi) * 16 + (lane & 15);
      u32x4 v;
#pragma unroll
      for (int u = 0; u < 4; ++u)
        v[u] = pack2bf(acc[pi][2*u] + bias_all[m8 + 2*u],
                       acc[pi][2*u+1] + bias_all[m8 + 2*u+1]);
      *(u32x4*)&QKV[p * 192 + m8] = v;
    }
  } else {                                               // gate preacts -> f32 [p][256]
#pragma unroll
    for (int pi = 0; pi < 4; ++pi) {
      int p = p0 + (pi0 + pi) * 16 + (lane & 15);
      f32x4 v0, v1;
#pragma unroll
      for (int u = 0; u < 4; ++u) {
        v0[u] = acc[pi][u]     + bias_all[m8 + u];
        v1[u] = acc[pi][4 + u] + bias_all[m8 + 4 + u];
      }
      *(f32x4*)&Gx[p * 256 + (m8 - 192)]     = v0;
      *(f32x4*)&Gx[p * 256 + (m8 - 192) + 4] = v1;
    }
  }
}

// ---------------- kernel 3: flash attention per (n, head, 64-query block) ----------------
// S'[d,q] = sum_c K[c,d] Q[c,q] (C-tile: M=d, N=q) -> online softmax over d
// A[c,q] += V[c,d] P[d,q]       (P reshaped in-register into B-fragment layout)
__global__ void __launch_bounds__(128) attn_kernel(
    const unsigned short* QKV, unsigned short* Abuf)
{
  __shared__ __align__(16) unsigned short sq[64 * LDST];
  __shared__ __align__(16) unsigned short sk[32 * LDST];
  __shared__ __align__(16) unsigned short sv[16 * LDST];
  const int tid = threadIdx.x, lane = tid & 31, wave = tid >> 5;
  const int q0 = blockIdx.x * 64, g = blockIdx.y, n = blockIdx.z;
  const int hlf = (lane >> 4) & 1;
  const int n2304 = n * 2304;
  const u32x4 zv = {0, 0, 0, 0};

  { // Q tile [q][c], c padded 16->32 with zeros; one b128 load per thread
    int ql = tid >> 1, hs = tid & 1;
    *(u32x4*)&sq[ql * LDST + hs * 8] =
        *(const u32x4*)&QKV[(n2304 + q0 + ql) * 192 + g * 16 + hs * 8];
    *(u32x4*)&sq[ql * LDST + 16 + hs * 8] = zv;
  }
  __syncthreads();
  v16us bq = ld_frag_b(sq, wave * 16, lane);             // persistent Q fragment

  float mrun = -3e38f, lrun = 0.f;
  v8f acc  = {0,0,0,0,0,0,0,0};
  v8f zero = acc;

  for (int dc = 0; dc < 67; ++dc) {                      // D = 2116, chunks of 32
    int dbase = dc * 32;
    { // K^T tile [d][c]: 32 rows x 2 halves + zero upper halves (128 tasks)
      int dl = tid >> 2, quad = tid & 3;
      int d = dbase + dl;
      if (quad < 2) {
        u32x4 v = zv;
        if (d < 2116) {
          int dy = d / 46, pos = (dy + 1) * 48 + (d - dy * 46 + 1);
          v = *(const u32x4*)&QKV[(n2304 + pos) * 192 + 64 + g * 16 + quad * 8];
        }
        *(u32x4*)&sk[dl * LDST + quad * 8] = v;
      } else {
        *(u32x4*)&sk[dl * LDST + 16 + (quad - 2) * 8] = zv;
      }
    }
    { // V tile [c][d]: batched strided gather, b64 stores (128 tasks)
      int c = tid >> 3, dg = tid & 7;
      unsigned short t[4];
#pragma unroll
      for (int u = 0; u < 4; ++u) {
        int d = dbase + dg * 4 + u;
        unsigned short vv = 0;
        if (d < 2116) {
          int dy = d / 46, pos = (dy + 1) * 48 + (d - dy * 46 + 1);
          vv = QKV[(n2304 + pos) * 192 + 128 + g * 16 + c];
        }
        t[u] = vv;
      }
      u32x2 w;
      w[0] = (unsigned int)t[0] | ((unsigned int)t[1] << 16);
      w[1] = (unsigned int)t[2] | ((unsigned int)t[3] << 16);
      *(u32x2*)&sv[c * LDST + dg * 4] = w;
    }
    __syncthreads();
    v16us a0 = ld_frag_a(sk, (lane & 15), lane);         // d 0..15
    v16us a1 = ld_frag_a(sk, 16 + (lane & 15), lane);    // d 16..31
    v8f S0 = wmma_bf16(a0, bq, zero);
    v8f S1 = wmma_bf16(a1, bq, zero);

    float p0v[8], p1v[8], cmax = -3e38f;
#pragma unroll
    for (int j = 0; j < 8; ++j) {                        // mask padded d, chunk max
      int d0 = dbase + hlf * 8 + j;
      int d1 = dbase + 16 + hlf * 8 + j;
      p0v[j] = (d0 < 2116) ? S0[j] : -3e38f;
      p1v[j] = (d1 < 2116) ? S1[j] : -3e38f;
      cmax = fmaxf(cmax, fmaxf(p0v[j], p1v[j]));
    }
    cmax = fmaxf(cmax, __shfl_xor(cmax, 16, 32));        // half-wave pair shares q
    float mnew = fmaxf(mrun, cmax);
    float scale = __expf(mrun - mnew);
    float csum = 0.f;
#pragma unroll
    for (int j = 0; j < 8; ++j) {
      p0v[j] = __expf(p0v[j] - mnew);
      p1v[j] = __expf(p1v[j] - mnew);
      csum += p0v[j] + p1v[j];
    }
    csum += __shfl_xor(csum, 16, 32);
    lrun = lrun * scale + csum;
    mrun = mnew;
#pragma unroll
    for (int j = 0; j < 8; ++j) acc[j] *= scale;

    v16us pb;                                            // P -> B-fragment (K=d, N=q)
#pragma unroll
    for (int j = 0; j < 8; ++j) {
      float x0 = __shfl_xor(p0v[j], 16, 32);
      float x1 = __shfl_xor(p1v[j], 16, 32);
      pb[j]     = f2bf(hlf ? x1 : p0v[j]);               // K = 16h + j
      pb[8 + j] = f2bf(hlf ? p1v[j] : x0);               // K = 16h + 8 + j
    }
    v16us av = ld_frag_a(sv, (lane & 15), lane);         // V fragment (M=c, K=d)
    acc = wmma_bf16(av, pb, acc);
    __syncthreads();
  }
  float inv = 1.0f / lrun;
  { // packed b128 epilogue: Abuf [p][64]
    int q = q0 + wave * 16 + (lane & 15);
    u32x4 v;
#pragma unroll
    for (int u = 0; u < 4; ++u)
      v[u] = pack2bf(acc[2*u] * inv, acc[2*u+1] * inv);
    *(u32x4*)&Abuf[(n2304 + q) * 64 + g * 16 + hlf * 8] = v;
  }
}

// ---------------- kernel 4: gate GEMM (256x64x9216) + LSTM pointwise ----------------
__global__ void __launch_bounds__(256) gates_lstm_kernel(
    const unsigned short* Wa, const unsigned short* Abuf, const float* Gx,
    const float* c_in, float* h_out)
{
  extern __shared__ unsigned char smem[];
  unsigned short* sa = (unsigned short*)smem;            // 256*LDST ushorts
  unsigned short* sb = sa + 256 * LDST;                  // 32*LDST ushorts
  float* gbuf = (float*)smem;                            // aliases (post-GEMM)
  const int tid = threadIdx.x, lane = tid & 31, wave = tid >> 5;
  const int p0 = blockIdx.x * 32;                        // 2304 % 32 == 0
  const int hlf = (lane >> 4) & 1;

  v8f z8 = {0,0,0,0,0,0,0,0};
  v8f acc[2][2] = {{z8, z8}, {z8, z8}};

  for (int kc = 0; kc < 2; ++kc) {                       // K = 64
#if HAVE_TDM
    if (wave == 0)                                       // async weight tile DMA
      tdm_load_2d((unsigned)(uintptr_t)sa, (const void*)&Wa[kc * 32],
                  64, 256, 32, 256, 64);
#else
    for (int e = tid; e < 1024; e += 256) {
      int m = e >> 2, kg = e & 3;
      *(u32x4*)&sa[m * LDST + kg * 8] =
          *(const u32x4*)&Wa[m * 64 + kc * 32 + kg * 8];
    }
#endif
    if (tid < 128) {
      int col = tid >> 2, kg = tid & 3;
      *(u32x4*)&sb[col * LDST + kg * 8] =
          *(const u32x4*)&Abuf[(p0 + col) * 64 + kc * 32 + kg * 8];
    }
#if HAVE_TDM
    if (wave == 0) __builtin_amdgcn_s_wait_tensorcnt(0);
#endif
    __syncthreads();
#pragma unroll
    for (int t = 0; t < 2; ++t) {
      v16us af = ld_frag_a(sa, (wave * 2 + t) * 16 + (lane & 15), lane);
#pragma unroll
      for (int pi = 0; pi < 2; ++pi) {
        v16us bf = ld_frag_b(sb, pi * 16, lane);
        acc[t][pi] = wmma_bf16(af, bf, acc[t][pi]);
      }
    }
    __syncthreads();
  }
  // activations -> LDS gate buffer [256][36]
#pragma unroll
  for (int t = 0; t < 2; ++t) {
#pragma unroll
    for (int pi = 0; pi < 2; ++pi) {
      int mrow = (wave * 2 + t) * 16 + hlf * 8;
      int col = pi * 16 + (lane & 15);
      f32x4 g0 = *(const f32x4*)&Gx[(p0 + col) * 256 + mrow];
      f32x4 g1 = *(const f32x4*)&Gx[(p0 + col) * 256 + mrow + 4];
#pragma unroll
      for (int j = 0; j < 8; ++j) {
        int m = mrow + j;
        float v = acc[t][pi][j] + (j < 4 ? g0[j] : g1[j - 4]);
        int gate = m >> 6;                               // 0:i 1:f 2:g 3:o
        v = (gate == 2) ? tanhf(v) : 1.0f / (1.0f + __expf(-v));
        gbuf[m * 36 + col] = v;
      }
    }
  }
  __syncthreads();
  for (int e = tid; e < 64 * 32; e += 256) {
    int r = e >> 5, col = e & 31;
    int p = p0 + col;
    int n = p / 2304, pos = p - n * 2304;
    float iv = gbuf[r * 36 + col];
    float fv = gbuf[(64 + r) * 36 + col];
    float gv = gbuf[(128 + r) * 36 + col];
    float ov = gbuf[(192 + r) * 36 + col];
    float cold = c_in[(n * 64 + r) * 2304 + pos];
    float cn = fv * cold + iv * gv;
    h_out[(n * 64 + r) * 2304 + pos] = ov * tanhf(cn);
  }
}

// ---------------- host orchestration ----------------
extern "C" void kernel_launch(void* const* d_in, const int* in_sizes, int n_in,
                              void* d_out, int out_size, void* d_ws, size_t ws_size,
                              hipStream_t stream) {
  (void)in_sizes; (void)n_in; (void)out_size; (void)ws_size;
  const float* x_in = (const float*)d_in[0];
  const float* h    = (const float*)d_in[1];
  const float* c    = (const float*)d_in[2];
  const float* W_x  = (const float*)d_in[3];
  const float* b_x  = (const float*)d_in[4];
  const float* W_q  = (const float*)d_in[5];
  const float* W_k  = (const float*)d_in[6];
  const float* W_v  = (const float*)d_in[7];
  const float* b_v  = (const float*)d_in[8];
  const float* Wi_a = (const float*)d_in[9];
  const float* Wi_x = (const float*)d_in[10];
  const float* b_i  = (const float*)d_in[11];
  const float* Wf_a = (const float*)d_in[12];
  const float* Wf_x = (const float*)d_in[13];
  const float* b_f  = (const float*)d_in[14];
  const float* Wg_a = (const float*)d_in[15];
  const float* Wg_x = (const float*)d_in[16];
  const float* b_g  = (const float*)d_in[17];
  const float* Wo_a = (const float*)d_in[18];
  const float* Wo_x = (const float*)d_in[19];
  const float* b_o  = (const float*)d_in[20];
  float* out = (float*)d_out;

  uint8_t* ws = (uint8_t*)d_ws;
  size_t off = 0;
  auto take = [&](size_t bytes) -> void* {
    void* p = ws + off;
    off += (bytes + 255) & ~(size_t)255;
    return p;
  };
  unsigned short* Wall = (unsigned short*)take(448UL * 1152 * 2);
  unsigned short* Wx   = (unsigned short*)take(64UL * 128 * 2);
  unsigned short* Wa   = (unsigned short*)take(256UL * 64 * 2);
  float*          bias = (float*)take(448UL * 4);
  unsigned short* xh   = (unsigned short*)take(9216UL * 128 * 2);
  unsigned short* QKV  = (unsigned short*)take(9216UL * 192 * 2);
  float*          Gx   = (float*)take(9216UL * 256 * 4);
  unsigned short* Abuf = (unsigned short*)take(9216UL * 64 * 2);

  prep_kernel<<<512, 256, 0, stream>>>(W_x, W_q, W_k, W_v, Wi_x, Wf_x, Wg_x, Wo_x,
                                       Wi_a, Wf_a, Wg_a, Wo_a, b_v, b_i, b_f, b_g, b_o,
                                       Wall, Wx, Wa, bias);
  proj_concat_kernel<<<72, 256, 0, stream>>>(x_in, h, Wx, b_x, xh);
  conv448_kernel<<<dim3(72, 7), 256, 0, stream>>>(xh, Wall, bias, QKV, Gx);
  attn_kernel<<<dim3(36, 4, 4), 128, 0, stream>>>(QKV, Abuf);
  gates_lstm_kernel<<<288, 256, 256 * 36 * 4, stream>>>(Wa, Abuf, Gx, c, out);
}
